// Interaction_35536559407622
// MI455X (gfx1250) — matrix operand order repaired and checked by reading, not compiled, not run on hardware
//
#include <hip/hip_runtime.h>
#include <hip/hip_bf16.h>
#include <math.h>

// ---------------------------------------------------------------------------
// CDNA5 (gfx1250) WMMA helpers: 16x16x32 bf16 -> f32
// ---------------------------------------------------------------------------
typedef __attribute__((ext_vector_type(16))) __bf16 v16bf;
typedef __attribute__((ext_vector_type(8)))  float  v8f;

#define WMMA_BF16(a, b, c) \
  __builtin_amdgcn_wmma_f32_16x16x32_bf16(false, (a), false, (b), (short)0, (c), false, false)

// Fast sigmoid: v_exp_f32 + v_rcp_f32 (avoids the IEEE div-scale sequence).
static __device__ __forceinline__ float sigmoidf_(float x) {
  return __builtin_amdgcn_rcpf(1.0f + __expf(-x));
}

// ---------------------------------------------------------------------------
// CDNA5 async global->LDS copy (ASYNCcnt path), 16 bytes per lane.
// GV addressing: LDS address from the vdst VGPR, global address from the 64-bit vaddr.
// LDS byte offset = low 32 bits of the generic shared-pointer address
// (ISA 10.2: LDS aperture maps addr[31:0] -> LDS address).
// ---------------------------------------------------------------------------
static __device__ __forceinline__ void async_copy_b128(void* lds_dst, const float* gsrc) {
  const unsigned ldsoff = (unsigned)(uintptr_t)lds_dst;
  asm volatile("global_load_async_to_lds_b128 %0, %1, off"
               :: "v"(ldsoff), "v"(gsrc) : "memory");
}
static __device__ __forceinline__ void wait_async0() {
  asm volatile("s_wait_asynccnt 0x0" ::: "memory");
}

// A-matrix 16x32 (MxK) fragment from row-major f32, leading dim `ld`.
// The 16 per-lane elements are the contiguous run
// k = kh..kh+15 (kh = 16 for lanes 16-31), in order.
// -> 4 x b128 loads (global or ds depending on pointer).
static __device__ __forceinline__ void load_a_frag(const float* src, int ld, v16bf& a) {
  const int lane = threadIdx.x & 31;
  const int m    = lane & 15;
  const int kh   = (lane & 16) ? 16 : 0;
  const float* p = src + (size_t)m * ld + kh;
  const float4 q0 = *(const float4*)(p);
  const float4 q1 = *(const float4*)(p + 4);
  const float4 q2 = *(const float4*)(p + 8);
  const float4 q3 = *(const float4*)(p + 12);
  a[0] = (__bf16)q0.x;  a[1] = (__bf16)q0.y;  a[2]  = (__bf16)q0.z;  a[3]  = (__bf16)q0.w;
  a[4] = (__bf16)q1.x;  a[5] = (__bf16)q1.y;  a[6]  = (__bf16)q1.z;  a[7]  = (__bf16)q1.w;
  a[8] = (__bf16)q2.x;  a[9] = (__bf16)q2.y;  a[10] = (__bf16)q2.z;  a[11] = (__bf16)q2.w;
  a[12] = (__bf16)q3.x; a[13] = (__bf16)q3.y; a[14] = (__bf16)q3.z;  a[15] = (__bf16)q3.w;
}

// B-matrix 32x16 (KxN) fragment, B[k][n] = src[k*ld + n].
// Lanes 0-15 hold K=0..15, lanes 16-31 hold K=16..31; col n = lane&15.
static __device__ __forceinline__ void load_b_frag(const float* __restrict__ src, int ld, v16bf& b) {
  const int lane = threadIdx.x & 31;
  const int n    = lane & 15;
  const int kb   = (lane & 16) ? 16 : 0;
  const float* p = src + (size_t)kb * ld + n;
#pragma unroll
  for (int i = 0; i < 16; ++i) b[i] = (__bf16)p[(size_t)i * ld];
}

// B-matrix 32x16 fragment from transposed source: B[k][n] = src[n*ld + k].
// Per lane this is 16 contiguous floats -> 4 x b128 loads.
static __device__ __forceinline__ void load_bT_frag(const float* __restrict__ src, int ld, v16bf& b) {
  const int lane = threadIdx.x & 31;
  const int n    = lane & 15;
  const int kb   = (lane & 16) ? 16 : 0;
  const float* p = src + (size_t)n * ld + kb;
  const float4 q0 = *(const float4*)(p);
  const float4 q1 = *(const float4*)(p + 4);
  const float4 q2 = *(const float4*)(p + 8);
  const float4 q3 = *(const float4*)(p + 12);
  b[0] = (__bf16)q0.x;  b[1] = (__bf16)q0.y;  b[2]  = (__bf16)q0.z;  b[3]  = (__bf16)q0.w;
  b[4] = (__bf16)q1.x;  b[5] = (__bf16)q1.y;  b[6]  = (__bf16)q1.z;  b[7]  = (__bf16)q1.w;
  b[8] = (__bf16)q2.x;  b[9] = (__bf16)q2.y;  b[10] = (__bf16)q2.z;  b[11] = (__bf16)q2.w;
  b[12] = (__bf16)q3.x; b[13] = (__bf16)q3.y; b[14] = (__bf16)q3.z;  b[15] = (__bf16)q3.w;
}

// Store a 16x16 f32 C fragment to row-major dst (leading dim ldc).
static __device__ __forceinline__ void store_c_frag(float* __restrict__ dst, int ldc, const v8f& c,
                                                    const float bv, const int act) {
  const int lane = threadIdx.x & 31;
  const int n    = lane & 15;
  const int mb   = (lane & 16) ? 8 : 0;
#pragma unroll
  for (int r = 0; r < 8; ++r) {
    float v = c[r] + bv;
    if (act == 1) v = sigmoidf_(v);
    dst[(size_t)(mb + r) * ldc + n] = v;
  }
}

// ---------------------------------------------------------------------------
// Generic WMMA GEMM:  Y[M,N] = act(X[M,K] @ W[K,N] (+ bias))
// act: 0 = none(+bias), 1 = sigmoid.
// The 16x32 A tile is staged in LDS once per block via async DMA
// (double-buffered, overlaps the WMMA of the current step); all 4 waves
// share it. Each wave computes a 16x32 output strip (2 accumulators).
// grid = (M/16, N/128), block = 128 (4 waves).  N mult of 128, K mult of 32.
// ---------------------------------------------------------------------------
__global__ __launch_bounds__(128)
void gemm_act_kernel(const float* __restrict__ X, const float* __restrict__ W,
                     const float* __restrict__ bias, float* __restrict__ Y,
                     int M, int N, int K, int act) {
  __shared__ float As[2][16][32];  // 4 KB double buffer
  const int wave = threadIdx.x >> 5;
  const int tm   = blockIdx.x;
  const int tp   = blockIdx.y * 4 + wave;   // 16x32 strip index
  const int tn0  = tp * 2, tn1 = tn0 + 1;

  const float* Xt = X + (size_t)tm * 16 * K;
  // Per-thread slice of the 16x32 A tile: one b128 async copy each.
  const int arow = threadIdx.x >> 3;        // 0..15
  const int acol = (threadIdx.x & 7) * 4;   // 0,4,...,28

  async_copy_b128(&As[0][arow][acol], Xt + (size_t)arow * K + acol);

  v8f acc0 = {}, acc1 = {};
  const int steps = K / 32;
  for (int s = 0; s < steps; ++s) {
    const int cur = s & 1;
    wait_async0();        // our async write to As[cur] has landed
    __syncthreads();      // ...and everyone else's too
    if (s + 1 < steps) {
      async_copy_b128(&As[1 - cur][arow][acol],
                      Xt + (size_t)arow * K + (s + 1) * 32 + acol);
      __builtin_prefetch(W + (size_t)(s + 1) * 32 * N + tn0 * 16, 0, 3);
    }
    v16bf a;  load_a_frag(&As[cur][0][0], 32, a);   // ds_load_b128 path
    v16bf b0; load_b_frag(W + (size_t)s * 32 * N + tn0 * 16, N, b0);
    acc0 = WMMA_BF16(a, b0, acc0);
    v16bf b1; load_b_frag(W + (size_t)s * 32 * N + tn1 * 16, N, b1);
    acc1 = WMMA_BF16(a, b1, acc1);
    __syncthreads();      // readers done before next iteration's DMA overwrite
  }

  const int n  = threadIdx.x & 15;
  const float bv0 = bias ? bias[tn0 * 16 + n] : 0.0f;
  const float bv1 = bias ? bias[tn1 * 16 + n] : 0.0f;
  float* dst = Y + (size_t)tm * 16 * N;
  store_c_frag(dst + tn0 * 16, N, acc0, bv0, act);
  store_c_frag(dst + tn1 * 16, N, acc1, bv1, act);
}

// ---------------------------------------------------------------------------
// Fused flash-style attention for one (b, head, 16-query tile):
//   S = softmax( (Q_tile @ K^T) / sqrt(H) );  O = S @ K
// qx,kx layout: [NH, B, L, H] row-major.  out layout: [B, L, NH*H].
// grid = (L/16, NH*B), block = 128, dyn-LDS = 16*L*4 bytes (64 KB).
// ---------------------------------------------------------------------------
__global__ __launch_bounds__(128)
void attn_kernel(const float* __restrict__ Qx, const float* __restrict__ Kx,
                 float* __restrict__ out, int B, int L, int H) {
  extern __shared__ float S[];  // [16][L]
  const int qt   = blockIdx.x;
  const int nb   = blockIdx.y;        // head*B + b
  const int head = nb / B;
  const int b    = nb % B;
  const int wave = threadIdx.x >> 5;
  const float scale = __builtin_amdgcn_rcpf(sqrtf((float)H));

  const float* Q  = Qx + (size_t)nb * L * H + (size_t)qt * 16 * H;
  const float* Km = Kx + (size_t)nb * L * H;

  // Preload A-fragments of the 16x128 Q tile (H/32 = 4 K-steps).
  v16bf aq[4];
#pragma unroll
  for (int hk = 0; hk < 4; ++hk) load_a_frag(Q + hk * 32, H, aq[hk]);

  // ---- Phase 1: S[16, L] = scale * Q_tile @ K^T ----
  const int nkt = L / 16;
  for (int kt = wave; kt < nkt; kt += 4) {
    __builtin_prefetch(Km + (size_t)(kt + 4) * 16 * H, 0, 3);
    v8f acc = {};
#pragma unroll
    for (int hk = 0; hk < 4; ++hk) {
      v16bf bf; load_bT_frag(Km + (size_t)kt * 16 * H + hk * 32, H, bf);
      acc = WMMA_BF16(aq[hk], bf, acc);
    }
    const int lane = threadIdx.x & 31;
    const int nn   = lane & 15;
    const int mb   = (lane & 16) ? 8 : 0;
#pragma unroll
    for (int r = 0; r < 8; ++r) S[(size_t)(mb + r) * L + kt * 16 + nn] = acc[r] * scale;
  }
  __syncthreads();

  // ---- Phase 2: row softmax over L (8 threads per row, 16 rows) ----
  {
    const int row = threadIdx.x >> 3;  // 0..15
    const int sub = threadIdx.x & 7;   // 0..7 (contiguous lanes in a wave)
    float* Sr = S + (size_t)row * L;
    float mx = -1e30f;
    for (int c = sub; c < L; c += 8) mx = fmaxf(mx, Sr[c]);
#pragma unroll
    for (int off = 4; off; off >>= 1) mx = fmaxf(mx, __shfl_xor(mx, off, 8));
    float sum = 0.0f;
    for (int c = sub; c < L; c += 8) {
      const float e = __expf(Sr[c] - mx);
      Sr[c] = e;
      sum += e;
    }
#pragma unroll
    for (int off = 4; off; off >>= 1) sum += __shfl_xor(sum, off, 8);
    const float inv = __builtin_amdgcn_rcpf(sum);
    for (int c = sub; c < L; c += 8) Sr[c] *= inv;
  }
  __syncthreads();

  // ---- Phase 3: O[16, H] = P[16, L] @ K[L, H]; wave handles n-tiles w, w+4 ----
  v8f o0 = {}, o1 = {};
  const int nt0 = wave, nt1 = wave + 4;  // H/16 = 8 tiles
  for (int kk = 0; kk < L / 32; ++kk) {
    v16bf ap; load_a_frag(S + kk * 32, L, ap);   // ds_load_b128 path
    v16bf b0; load_b_frag(Km + (size_t)kk * 32 * H + nt0 * 16, H, b0);
    o0 = WMMA_BF16(ap, b0, o0);
    v16bf b1; load_b_frag(Km + (size_t)kk * 32 * H + nt1 * 16, H, b1);
    o1 = WMMA_BF16(ap, b1, o1);
  }

  // Store into concatenated-head layout [B, L, NH*H]
  const int NHH  = 2 * H;
  float* dst = out + ((size_t)b * L + qt * 16) * NHH + head * H;
  store_c_frag(dst + nt0 * 16, NHH, o0, 0.0f, 0);
  store_c_frag(dst + nt1 * 16, NHH, o1, 0.0f, 0);
}

// ---------------------------------------------------------------------------
// M-attention tail kernels
// ---------------------------------------------------------------------------
// logits[r] = sigmoid( dot(kxm[r]*qxm[r], Wm) + bm ) / sqrt(HM)
__global__ void mlogit_kernel(const float* __restrict__ kxm, const float* __restrict__ qxm,
                              const float* __restrict__ Wm, const float* __restrict__ bm,
                              float* __restrict__ logits, int rows, int E) {
  const int r = blockIdx.x * blockDim.x + threadIdx.x;
  if (r >= rows) return;
  const float4* kk = (const float4*)(kxm + (size_t)r * E);
  const float4* qq = (const float4*)(qxm + (size_t)r * E);
  const float4* wm = (const float4*)Wm;
  float acc = 0.0f;
  for (int e = 0; e < E / 4; ++e) {
    const float4 k4 = kk[e], q4 = qq[e], w4 = wm[e];
    acc += k4.x * q4.x * w4.x + k4.y * q4.y * w4.y + k4.z * q4.z * w4.z + k4.w * q4.w * w4.w;
  }
  logits[r] = sigmoidf_(acc + bm[0]) * __builtin_amdgcn_rcpf(sqrtf((float)E));
}

// per-batch softmax over L
__global__ __launch_bounds__(256)
void msoftmax_kernel(const float* __restrict__ logits, float* __restrict__ probs, int L) {
  __shared__ float red[256];
  const int b = blockIdx.x;
  const float* src = logits + (size_t)b * L;
  float mx = -1e30f;
  for (int i = threadIdx.x; i < L; i += 256) mx = fmaxf(mx, src[i]);
  red[threadIdx.x] = mx; __syncthreads();
  for (int s = 128; s; s >>= 1) {
    if (threadIdx.x < s) red[threadIdx.x] = fmaxf(red[threadIdx.x], red[threadIdx.x + s]);
    __syncthreads();
  }
  mx = red[0]; __syncthreads();
  float sum = 0.0f;
  for (int i = threadIdx.x; i < L; i += 256) sum += __expf(src[i] - mx);
  red[threadIdx.x] = sum; __syncthreads();
  for (int s = 128; s; s >>= 1) {
    if (threadIdx.x < s) red[threadIdx.x] += red[threadIdx.x + s];
    __syncthreads();
  }
  const float inv = __builtin_amdgcn_rcpf(red[0]);
  for (int i = threadIdx.x; i < L; i += 256) probs[(size_t)b * L + i] = __expf(src[i] - mx) * inv;
}

// out[b, e] = sum_l probs[b,l] * feat[b,l,e]
__global__ __launch_bounds__(256)
void wsum_kernel(const float* __restrict__ probs, const float* __restrict__ feat,
                 float* __restrict__ out, int L, int E) {
  const int b = blockIdx.x;
  const int e = threadIdx.x;
  const float* f = feat + (size_t)b * L * E + e;
  const float* p = probs + (size_t)b * L;
  float acc = 0.0f;
  for (int l = 0; l < L; ++l) acc += p[l] * f[(size_t)l * E];
  out[(size_t)b * E + e] = acc;
}

// ---------------------------------------------------------------------------
// Host orchestration
// ---------------------------------------------------------------------------
extern "C" void kernel_launch(void* const* d_in, const int* in_sizes, int n_in,
                              void* d_out, int out_size, void* d_ws, size_t ws_size,
                              hipStream_t stream) {
  (void)in_sizes; (void)n_in; (void)out_size; (void)ws_size;
  constexpr int B = 16, L = 1024, E = 256, NH = 2, H = 128, OUT = 256;
  constexpr int BL = B * L;                 // 16384
  constexpr size_t F = (size_t)BL * E;      // 4,194,304 floats per [B,L,256] buffer

  const float* x_d   = (const float*)d_in[0];
  const float* x_p   = (const float*)d_in[1];
  const float* wkx_d = (const float*)d_in[2];
  const float* wqx_d = (const float*)d_in[3];
  const float* Wp_d  = (const float*)d_in[4];
  const float* bp_d  = (const float*)d_in[5];
  const float* wkx_p = (const float*)d_in[6];
  const float* wqx_p = (const float*)d_in[7];
  const float* Wp_p  = (const float*)d_in[8];
  const float* bp_p  = (const float*)d_in[9];
  const float* wkx_m = (const float*)d_in[10];
  const float* wqx_m = (const float*)d_in[11];
  const float* Wm    = (const float*)d_in[12];
  const float* bm    = (const float*)d_in[13];
  float* out = (float*)d_out;

  // Workspace carve-up (floats). Total ~10*F + 32K ≈ 168 MB.
  float* ws     = (float*)d_ws;
  float* kx_d   = ws + 0 * F;   // [NH,B,L,H]
  float* qx_d   = ws + 1 * F;
  float* kx_p   = ws + 2 * F;
  float* qx_p   = ws + 3 * F;
  float* attn_d = ws + 4 * F;   // [B,L,NH*H]
  float* attn_p = ws + 5 * F;
  float* d_feat = ws + 6 * F;   // [B,L,OUT]
  float* p_feat = ws + 7 * F;
  float* kxm    = ws + 8 * F;   // [B,L,HM]
  float* qxm    = ws + 9 * F;
  float* logit  = ws + 10 * F;  // [B,L]
  float* probs  = logit + BL;

  const size_t headW = (size_t)E * H;        // weight stride per head
  const size_t headY = (size_t)B * L * H;    // output stride per head

  const dim3 blk(128);
  const dim3 gH(BL / 16, H / 128);           // N=128 GEMMs
  const dim3 gO(BL / 16, OUT / 128);         // N=256 GEMMs

  // 1) sigmoid projections for both branches, both heads, k & q
  gemm_act_kernel<<<gH, blk, 0, stream>>>(x_d, wkx_d,          nullptr, kx_d,          BL, H, E, 1);
  gemm_act_kernel<<<gH, blk, 0, stream>>>(x_d, wkx_d + headW,  nullptr, kx_d + headY,  BL, H, E, 1);
  gemm_act_kernel<<<gH, blk, 0, stream>>>(x_d, wqx_d,          nullptr, qx_d,          BL, H, E, 1);
  gemm_act_kernel<<<gH, blk, 0, stream>>>(x_d, wqx_d + headW,  nullptr, qx_d + headY,  BL, H, E, 1);
  gemm_act_kernel<<<gH, blk, 0, stream>>>(x_p, wkx_p,          nullptr, kx_p,          BL, H, E, 1);
  gemm_act_kernel<<<gH, blk, 0, stream>>>(x_p, wkx_p + headW,  nullptr, kx_p + headY,  BL, H, E, 1);
  gemm_act_kernel<<<gH, blk, 0, stream>>>(x_p, wqx_p,          nullptr, qx_p,          BL, H, E, 1);
  gemm_act_kernel<<<gH, blk, 0, stream>>>(x_p, wqx_p + headW,  nullptr, qx_p + headY,  BL, H, E, 1);

  // 2) fused attention (scores live in LDS only)
  const dim3 gA(L / 16, NH * B);
  const size_t smem = 16 * (size_t)L * sizeof(float);  // 64 KB
  attn_kernel<<<gA, blk, smem, stream>>>(qx_d, kx_d, attn_d, B, L, H);
  attn_kernel<<<gA, blk, smem, stream>>>(qx_p, kx_p, attn_p, B, L, H);

  // 3) output projections (+bias)
  gemm_act_kernel<<<gO, blk, 0, stream>>>(attn_d, Wp_d, bp_d, d_feat, BL, OUT, NH * H, 0);
  gemm_act_kernel<<<gO, blk, 0, stream>>>(attn_p, Wp_p, bp_p, p_feat, BL, OUT, NH * H, 0);

  // 4) M-attention: sigmoid projections, gate, softmax over L, weighted sum of D
  gemm_act_kernel<<<gO, blk, 0, stream>>>(d_feat, wkx_m, nullptr, kxm, BL, E, OUT, 1);
  gemm_act_kernel<<<gO, blk, 0, stream>>>(p_feat, wqx_m, nullptr, qxm, BL, E, OUT, 1);
  mlogit_kernel<<<BL / 256, 256, 0, stream>>>(kxm, qxm, Wm, bm, logit, BL, E);
  msoftmax_kernel<<<B, 256, 0, stream>>>(logit, probs, L);
  wsum_kernel<<<B, 256, 0, stream>>>(probs, d_feat, out, L, E);
}
